// MoEResNetBKLayer_55594056680078
// MI455X (gfx1250) — matrix operand order repaired and checked by reading, not compiled, not run on hardware
//
#include <hip/hip_runtime.h>
#include <hip/hip_bf16.h>
#include <math.h>

// Problem constants (B, N, D, E, K=1, H) from the reference.
constexpr int Bq = 4, Nq = 2048, Dq = 1024, Eq = 4, Hq = 4096;
constexpr int Tq = Bq * Nq;          // 8192 tokens
constexpr int SA = 40;               // LDS row stride in halves (32 data + 8 pad = 80B, 16B aligned)

typedef _Float16 half4v  __attribute__((ext_vector_type(4)));
typedef _Float16 half8v  __attribute__((ext_vector_type(8)));
typedef _Float16 half16v __attribute__((ext_vector_type(16)));
typedef float    float8v __attribute__((ext_vector_type(8)));
typedef int      int4v   __attribute__((ext_vector_type(4)));

// ---------------- CDNA5 async-copy helpers (ASYNCcnt path) ----------------

__device__ __forceinline__ void async_copy16(const _Float16* g, _Float16* l) {
#if __has_builtin(__builtin_amdgcn_global_load_async_to_lds_b128)
  // Signature (from clang diagnostic): (v4i global*, v4i shared*, Ii offset, Ii cpol).
  // Address-space-qualified pointers built via integer casts (flat->global is
  // numerically identity; low 32 bits of a flat shared address = LDS offset).
  __builtin_amdgcn_global_load_async_to_lds_b128(
      (__attribute__((address_space(1))) int4v*)(unsigned long long)g,
      (__attribute__((address_space(3))) int4v*)(unsigned)(unsigned long long)l,
      0, 0);
#else
  // VDST = per-lane LDS byte offset, VADDR = 64-bit global address, GV mode.
  asm volatile("global_load_async_to_lds_b128 %0, %1, off"
               :: "v"((unsigned)(unsigned long long)l),
                  "v"((unsigned long long)g)
               : "memory");
#endif
}

template <int N>
__device__ __forceinline__ void wait_async() {
#if __has_builtin(__builtin_amdgcn_s_wait_asynccnt)
  __builtin_amdgcn_s_wait_asynccnt(N);
#else
  asm volatile("s_wait_asynccnt %0" :: "i"(N) : "memory");
#endif
}

// ---------------- small utility kernels ----------------

__global__ void k_init(int* ints) {
  if (threadIdx.x < 16) ints[threadIdx.x] = 0;
}

// cast x (T*D f32) -> f16, vectorized
__global__ __launch_bounds__(256) void k_cast_x(const float* __restrict__ x,
                                                _Float16* __restrict__ xh) {
  int i = blockIdx.x * blockDim.x + threadIdx.x;   // over T*D/4
  float4 v = ((const float4*)x)[i];
  half4v h = { (_Float16)v.x, (_Float16)v.y, (_Float16)v.z, (_Float16)v.w };
  ((half4v*)xh)[i] = h;
}

// per-expert tiled transpose + cast: src [E][R][C] f32 -> dst [E][C][R] f16
__global__ void k_transpose_cast(const float* __restrict__ src,
                                 _Float16* __restrict__ dst, int R, int C) {
  __shared__ float tile[32][33];
  int e = blockIdx.z;
  const float* s = src + (size_t)e * R * C;
  _Float16*    d = dst + (size_t)e * R * C;
  int c0 = blockIdx.x * 32, r0 = blockIdx.y * 32;
  int tx = threadIdx.x, ty = threadIdx.y;          // block (32,8)
#pragma unroll
  for (int i = 0; i < 32; i += 8)
    tile[ty + i][tx] = s[(size_t)(r0 + ty + i) * C + (c0 + tx)];
  __syncthreads();
#pragma unroll
  for (int i = 0; i < 32; i += 8)
    d[(size_t)(c0 + ty + i) * R + (r0 + tx)] = (_Float16)tile[tx][ty + i];
}

// router: one wave per token, argmax over E=4 logits (first-max wins, matching top_k ties)
__global__ __launch_bounds__(256) void k_router(const float* __restrict__ x,
                                                const float* __restrict__ rw,
                                                const float* __restrict__ rb,
                                                int* __restrict__ expert,
                                                int* __restrict__ ints) {
  int wave = threadIdx.x >> 5, lane = threadIdx.x & 31;
  int t = blockIdx.x * 8 + wave;
  const float* xr = x + (size_t)t * Dq;
  float a0 = 0.f, a1 = 0.f, a2 = 0.f, a3 = 0.f;
  for (int j = 0; j < Dq / 32; ++j) {
    int dd = lane + 32 * j;
    float xv = xr[dd];
    float4 w = ((const float4*)rw)[dd];              // router_w row d: 4 experts
    a0 += xv * w.x; a1 += xv * w.y; a2 += xv * w.z; a3 += xv * w.w;
  }
#pragma unroll
  for (int s = 16; s > 0; s >>= 1) {
    a0 += __shfl_xor(a0, s, 32);
    a1 += __shfl_xor(a1, s, 32);
    a2 += __shfl_xor(a2, s, 32);
    a3 += __shfl_xor(a3, s, 32);
  }
  if (lane == 0) {
    float l[4] = { a0 + rb[0], a1 + rb[1], a2 + rb[2], a3 + rb[3] };
    int best = 0; float bv = l[0];
#pragma unroll
    for (int e = 1; e < 4; ++e) if (l[e] > bv) { bv = l[e]; best = e; }
    expert[t] = best;
    atomicAdd(&ints[best], 1);                       // counts
  }
}

// ints layout: [0..3]=counts, [4..7]=cursors, [8..12]=offsets
__global__ void k_offsets(int* ints) {
  ints[8] = 0;
  for (int e = 0; e < 4; ++e) ints[9 + e] = ints[8 + e] + ints[e];
}

__global__ __launch_bounds__(256) void k_scatter(const int* __restrict__ expert,
                                                 int* __restrict__ ints,
                                                 int* __restrict__ perm) {
  int t = blockIdx.x * blockDim.x + threadIdx.x;
  if (t >= Tq) return;
  int e = expert[t];
  int pos = atomicAdd(&ints[4 + e], 1);
  perm[ints[8 + e] + pos] = t;
}

// ---------------- WMMA fragment loaders (ISA §7.12.2 layouts) ----------------

// A 16x32 f16: lanes 0-15 -> K{0..7,16..23}; lanes 16-31 -> K{8..15,24..31}
__device__ __forceinline__ half16v frag_a(const _Float16* base, int hi) {
  half8v lo = *(const half8v*)(base + hi * 8);
  half8v hh = *(const half8v*)(base + 16 + hi * 8);
  half16v r;
#pragma unroll
  for (int i = 0; i < 8; ++i) { r[i] = lo[i]; r[i + 8] = hh[i]; }
  return r;
}

// B 32x16 f16 (stored as N-major rows = columns): lanes 0-15 -> K{0..7,16..23}; 16-31 -> K{8..15,24..31}
__device__ __forceinline__ half16v frag_b(const _Float16* base, int hi) {
  half8v lo = *(const half8v*)(base + hi * 8);
  half8v hh = *(const half8v*)(base + 16 + hi * 8);
  half16v r;
#pragma unroll
  for (int i = 0; i < 8; ++i) { r[i] = lo[i]; r[i + 8] = hh[i]; }
  return r;
}

__device__ __forceinline__ float8v wmma_f16(half16v a, half16v b, float8v c) {
  return __builtin_amdgcn_wmma_f32_16x16x32_f16(false, a, false, b, (short)0, c,
                                                false, false);
}

// ---------------- GEMM1: h = gelu(x[perm] @ w1[e] + b1[e]) -> f16 compacted ----------------
// grid (ceil(T/128), H/64, E), block 128 (4 waves); wave computes 32x64 of a 128x64 tile.
// K staged 32-wide via async loads to LDS, double buffered (ASYNCcnt pipelining).

__global__ __launch_bounds__(128) void k_gemm1(const _Float16* __restrict__ xh,
                                               const _Float16* __restrict__ w1t,
                                               const float* __restrict__ b1,
                                               _Float16* __restrict__ hbuf,
                                               const int* __restrict__ perm,
                                               const int* __restrict__ ints) {
  __shared__ __align__(16) _Float16 As[2][128 * SA];
  __shared__ __align__(16) _Float16 Bs[2][64 * SA];
  const int e = blockIdx.z;
  const int off0 = ints[8 + e];
  const int count = ints[9 + e] - off0;
  const int m0 = blockIdx.x * 128;
  if (m0 >= count) return;                       // uniform per block
  const int n0 = blockIdx.y * 64;
  const int tid = threadIdx.x;
  const int wave = tid >> 5, lane = tid & 31;
  const int r = lane & 15, hi = lane >> 4;

  // per-thread async-copy assignments: A = 128 rows x 4 chunks (16B), B = 64 rows x 4 chunks
  const _Float16* agp[4]; int aoff[4];
#pragma unroll
  for (int j = 0; j < 4; ++j) {
    int c = tid + 128 * j;
    int row = c >> 2, ch = c & 3;
    int m = m0 + row; if (m >= count) m = count - 1;   // clamp; writes masked later
    agp[j]  = xh + (size_t)perm[off0 + m] * Dq + ch * 8;
    aoff[j] = row * SA + ch * 8;
  }
  const _Float16* wbase = w1t + ((size_t)e * Hq + n0) * Dq;   // [H][D] rows = out cols
  const _Float16* bgp[2]; int boff[2];
#pragma unroll
  for (int j = 0; j < 2; ++j) {
    int c = tid + 128 * j;
    int row = c >> 2, ch = c & 3;
    bgp[j]  = wbase + (size_t)row * Dq + ch * 8;
    boff[j] = row * SA + ch * 8;
  }

  auto issue = [&](int kk, int p) {
#pragma unroll
    for (int j = 0; j < 4; ++j) async_copy16(agp[j] + kk, &As[p][aoff[j]]);
#pragma unroll
    for (int j = 0; j < 2; ++j) async_copy16(bgp[j] + kk, &Bs[p][boff[j]]);
  };

  float8v acc[2][4] = {};
  int p = 0;
  issue(0, 0);
  for (int kk = 0; kk < Dq; kk += 32) {
    if (kk + 32 < Dq) { issue(kk + 32, p ^ 1); wait_async<6>(); }
    else              wait_async<0>();
    __syncthreads();                                  // tile k resident for all waves
    half16v a0 = frag_a(&As[p][(wave * 32 + r) * SA], hi);
    half16v a1 = frag_a(&As[p][(wave * 32 + 16 + r) * SA], hi);
#pragma unroll
    for (int nb = 0; nb < 4; ++nb) {
      half16v b = frag_b(&Bs[p][(nb * 16 + r) * SA], hi);
      acc[0][nb] = wmma_f16(a0, b, acc[0][nb]);
      acc[1][nb] = wmma_f16(a1, b, acc[1][nb]);
    }
    __syncthreads();                                  // done reading buf p
    p ^= 1;
  }
  // epilogue: C layout — VGPR i, lane l: row = (l>>4)*8+i, col = l&15
#pragma unroll
  for (int sm = 0; sm < 2; ++sm) {
#pragma unroll
    for (int nb = 0; nb < 4; ++nb) {
      int n = n0 + nb * 16 + r;
      float bias = b1[e * Hq + n];
#pragma unroll
      for (int i = 0; i < 8; ++i) {
        int mrow = m0 + wave * 32 + sm * 16 + hi * 8 + i;
        if (mrow < count) {
          float xv = acc[sm][nb][i] + bias;
          float g = 0.5f * xv * (1.0f + erff(xv * 0.70710678118654752f));  // exact gelu
          hbuf[(size_t)(off0 + mrow) * Hq + n] = (_Float16)g;
        }
      }
    }
  }
}

// ---------------- GEMM2: ffn[perm] = h @ w2[e] + b2[e] -> f32 scattered into d_out ----------------

__global__ __launch_bounds__(128) void k_gemm2(const _Float16* __restrict__ hbuf,
                                               const _Float16* __restrict__ w2t,
                                               const float* __restrict__ b2,
                                               float* __restrict__ outp,
                                               const int* __restrict__ perm,
                                               const int* __restrict__ ints) {
  __shared__ __align__(16) _Float16 As[2][128 * SA];
  __shared__ __align__(16) _Float16 Bs[2][64 * SA];
  const int e = blockIdx.z;
  const int off0 = ints[8 + e];
  const int count = ints[9 + e] - off0;
  const int m0 = blockIdx.x * 128;
  if (m0 >= count) return;
  const int n0 = blockIdx.y * 64;
  const int tid = threadIdx.x;
  const int wave = tid >> 5, lane = tid & 31;
  const int r = lane & 15, hi = lane >> 4;

  const _Float16* agp[4]; int aoff[4];
#pragma unroll
  for (int j = 0; j < 4; ++j) {
    int c = tid + 128 * j;
    int row = c >> 2, ch = c & 3;
    int m = m0 + row; if (m >= count) m = count - 1;
    agp[j]  = hbuf + (size_t)(off0 + m) * Hq + ch * 8;   // compacted, no gather
    aoff[j] = row * SA + ch * 8;
  }
  const _Float16* wbase = w2t + ((size_t)e * Dq + n0) * Hq;   // [D][H]
  const _Float16* bgp[2]; int boff[2];
#pragma unroll
  for (int j = 0; j < 2; ++j) {
    int c = tid + 128 * j;
    int row = c >> 2, ch = c & 3;
    bgp[j]  = wbase + (size_t)row * Hq + ch * 8;
    boff[j] = row * SA + ch * 8;
  }

  auto issue = [&](int kk, int p) {
#pragma unroll
    for (int j = 0; j < 4; ++j) async_copy16(agp[j] + kk, &As[p][aoff[j]]);
#pragma unroll
    for (int j = 0; j < 2; ++j) async_copy16(bgp[j] + kk, &Bs[p][boff[j]]);
  };

  float8v acc[2][4] = {};
  int p = 0;
  issue(0, 0);
  for (int kk = 0; kk < Hq; kk += 32) {
    if (kk + 32 < Hq) { issue(kk + 32, p ^ 1); wait_async<6>(); }
    else              wait_async<0>();
    __syncthreads();
    half16v a0 = frag_a(&As[p][(wave * 32 + r) * SA], hi);
    half16v a1 = frag_a(&As[p][(wave * 32 + 16 + r) * SA], hi);
#pragma unroll
    for (int nb = 0; nb < 4; ++nb) {
      half16v b = frag_b(&Bs[p][(nb * 16 + r) * SA], hi);
      acc[0][nb] = wmma_f16(a0, b, acc[0][nb]);
      acc[1][nb] = wmma_f16(a1, b, acc[1][nb]);
    }
    __syncthreads();
    p ^= 1;
  }
#pragma unroll
  for (int sm = 0; sm < 2; ++sm) {
#pragma unroll
    for (int nb = 0; nb < 4; ++nb) {
      int n = n0 + nb * 16 + r;
      float bias = b2[e * Dq + n];
#pragma unroll
      for (int i = 0; i < 8; ++i) {
        int mrow = m0 + wave * 32 + sm * 16 + hi * 8 + i;
        if (mrow < count) {
          int t = perm[off0 + mrow];
          outp[(size_t)t * Dq + n] = acc[sm][nb][i] + bias;
        }
      }
    }
  }
}

// ---------------- v = clip(ffn @ v_w + v_b, ±3): one wave per token ----------------

__global__ __launch_bounds__(256) void k_vproj(const float* __restrict__ ffn,
                                               const float* __restrict__ vw,
                                               const float* __restrict__ vb,
                                               float* __restrict__ v) {
  int wave = threadIdx.x >> 5, lane = threadIdx.x & 31;
  int t = blockIdx.x * 8 + wave;
  const float4* xr = (const float4*)(ffn + (size_t)t * Dq);
  const float4* wr = (const float4*)vw;
  float s = 0.f;
  for (int j = 0; j < Dq / 128; ++j) {
    int i = lane + 32 * j;
    float4 a = xr[i], b = wr[i];
    s += a.x * b.x + a.y * b.y + a.z * b.z + a.w * b.w;
  }
#pragma unroll
  for (int sh = 16; sh > 0; sh >>= 1) s += __shfl_xor(s, sh, 32);
  if (lane == 0) {
    float val = s + vb[0];
    v[t] = fminf(fmaxf(val, -3.0f), 3.0f);
  }
}

// ---------------- Green's function: serial L/R continued fractions + elementwise G ----------------
// d_i = (-2 + v_i) - 1j ; L_0=0, L_i = 1/(d_{i-1}-L_{i-1}) ; R_{N-1}=0, R_i = 1/(d_{i+1}-R_{i+1})
// G_i = 1/(d_i - L_i - R_i), feats clipped to ±10.

__global__ __launch_bounds__(256) void k_scan(const float* __restrict__ v,
                                              float2* __restrict__ Lb,
                                              float2* __restrict__ Rb,
                                              float* __restrict__ Gre,
                                              float* __restrict__ Gim) {
  int tid = threadIdx.x;
  if (tid < Bq) {
    int b = tid;
    float lr = 0.f, li = 0.f;
    Lb[b * Nq] = make_float2(0.f, 0.f);
    for (int i = 1; i < Nq; ++i) {
      float ar = (-2.f + v[b * Nq + i - 1]) - lr;
      float ai = -1.f - li;
      float inv = 1.f / (ar * ar + ai * ai);
      lr = ar * inv; li = -ai * inv;
      Lb[b * Nq + i] = make_float2(lr, li);
    }
  } else if (tid < 2 * Bq) {
    int b = tid - Bq;
    float rr = 0.f, ri = 0.f;
    Rb[b * Nq + Nq - 1] = make_float2(0.f, 0.f);
    for (int i = Nq - 2; i >= 0; --i) {
      float ar = (-2.f + v[b * Nq + i + 1]) - rr;
      float ai = -1.f - ri;
      float inv = 1.f / (ar * ar + ai * ai);
      rr = ar * inv; ri = -ai * inv;
      Rb[b * Nq + i] = make_float2(rr, ri);
    }
  }
  __syncthreads();
  for (int idx = tid; idx < Tq; idx += blockDim.x) {
    float ar = (-2.f + v[idx]) - Lb[idx].x - Rb[idx].x;
    float ai = -1.f - Lb[idx].y - Rb[idx].y;
    float inv = 1.f / (ar * ar + ai * ai);
    float gr = ar * inv, gi = -ai * inv;
    Gre[idx] = fminf(fmaxf(gr, -10.f), 10.f);
    Gim[idx] = fminf(fmaxf(gi, -10.f), 10.f);
  }
}

// ---------------- out = ffn + bk*(G.re*out_w0 + G.im*out_w1 + out_b), in place ----------------

__global__ __launch_bounds__(256) void k_final(float* __restrict__ out,
                                               const float* __restrict__ Gre,
                                               const float* __restrict__ Gim,
                                               const float* __restrict__ ow,
                                               const float* __restrict__ ob,
                                               const float* __restrict__ bk) {
  int i = blockIdx.x * blockDim.x + threadIdx.x;   // over T*D/4
  int t = i / (Dq / 4);
  int dq = i % (Dq / 4);
  float scale = bk[0];
  float gr = Gre[t], gi = Gim[t];
  float4 w0 = ((const float4*)ow)[dq];
  float4 w1 = ((const float4*)(ow + Dq))[dq];
  float4 b  = ((const float4*)ob)[dq];
  float4 o  = ((float4*)out)[i];
  o.x += scale * (gr * w0.x + gi * w1.x + b.x);
  o.y += scale * (gr * w0.y + gi * w1.y + b.y);
  o.z += scale * (gr * w0.z + gi * w1.z + b.z);
  o.w += scale * (gr * w0.w + gi * w1.w + b.w);
  ((float4*)out)[i] = o;
}

// ---------------- host launcher ----------------

extern "C" void kernel_launch(void* const* d_in, const int* in_sizes, int n_in,
                              void* d_out, int out_size, void* d_ws, size_t ws_size,
                              hipStream_t stream) {
  const float* x  = (const float*)d_in[0];
  const float* rw = (const float*)d_in[1];
  const float* rb = (const float*)d_in[2];
  const float* w1 = (const float*)d_in[3];
  const float* b1 = (const float*)d_in[4];
  const float* w2 = (const float*)d_in[5];
  const float* b2 = (const float*)d_in[6];
  const float* vw = (const float*)d_in[7];
  const float* vb = (const float*)d_in[8];
  const float* ow = (const float*)d_in[9];
  const float* ob = (const float*)d_in[10];
  const float* bk = (const float*)d_in[11];
  float* out = (float*)d_out;

  char* ws = (char*)d_ws;
  size_t o = 0;
  auto alloc = [&](size_t bytes) -> char* {
    char* p = ws + o;
    o = (o + bytes + 255) & ~(size_t)255;
    return p;
  };
  _Float16* xh   = (_Float16*)alloc((size_t)Tq * Dq * 2);
  _Float16* w1t  = (_Float16*)alloc((size_t)Eq * Dq * Hq * 2);
  _Float16* w2t  = (_Float16*)alloc((size_t)Eq * Dq * Hq * 2);
  _Float16* hbuf = (_Float16*)alloc((size_t)Tq * Hq * 2);
  int*      ints = (int*)alloc(64);
  int*    expert = (int*)alloc((size_t)Tq * 4);
  int*      perm = (int*)alloc((size_t)Tq * 4);
  float*       v = (float*)alloc((size_t)Tq * 4);
  float2*     Lb = (float2*)alloc((size_t)Tq * 8);
  float2*     Rb = (float2*)alloc((size_t)Tq * 8);
  float*     Gre = (float*)alloc((size_t)Tq * 4);
  float*     Gim = (float*)alloc((size_t)Tq * 4);

  hipLaunchKernelGGL(k_init, dim3(1), dim3(32), 0, stream, ints);
  hipLaunchKernelGGL(k_cast_x, dim3(Tq * Dq / 4 / 256), dim3(256), 0, stream, x, xh);
  hipLaunchKernelGGL(k_transpose_cast, dim3(Hq / 32, Dq / 32, Eq), dim3(32, 8), 0, stream,
                     w1, w1t, Dq, Hq);   // [E][D][H] -> [E][H][D]
  hipLaunchKernelGGL(k_transpose_cast, dim3(Dq / 32, Hq / 32, Eq), dim3(32, 8), 0, stream,
                     w2, w2t, Hq, Dq);   // [E][H][D] -> [E][D][H]
  hipLaunchKernelGGL(k_router, dim3(Tq / 8), dim3(256), 0, stream, x, rw, rb, expert, ints);
  hipLaunchKernelGGL(k_offsets, dim3(1), dim3(1), 0, stream, ints);
  hipLaunchKernelGGL(k_scatter, dim3(Tq / 256), dim3(256), 0, stream, expert, ints, perm);
  hipLaunchKernelGGL(k_gemm1, dim3(Tq / 128, Hq / 64, Eq), dim3(128), 0, stream,
                     xh, w1t, b1, hbuf, perm, ints);
  hipLaunchKernelGGL(k_gemm2, dim3(Tq / 128, Dq / 64, Eq), dim3(128), 0, stream,
                     hbuf, w2t, b2, out, perm, ints);
  hipLaunchKernelGGL(k_vproj, dim3(Tq / 8), dim3(256), 0, stream, out, vw, vb, v);
  hipLaunchKernelGGL(k_scan, dim3(1), dim3(256), 0, stream, v, Lb, Rb, Gre, Gim);
  hipLaunchKernelGGL(k_final, dim3(Tq * Dq / 4 / 256), dim3(256), 0, stream,
                     out, Gre, Gim, ow, ob, bk);
}
